// RMCNN_44753559224838
// MI455X (gfx1250) — matrix operand order repaired and checked by reading, not compiled, not run on hardware
//
#include <hip/hip_runtime.h>
#include <hip/hip_bf16.h>
#include <math.h>

// ---------------------------------------------------------------------------
// MI455X (gfx1250) implementation.
// Roofline: x = 202 MB -> ~9us HBM floor @23.3TB/s; conv is 21 GFLOP ->
// must use v_wmma_f32_16x16x32_f16 (f16 in / f32 acc), not VALU.
// All GEMM-shaped math (conv-im2col, covariance, bimap) runs on WMMA.
// The raw-x stream (dominant HBM traffic) is staged with the CDNA5 async
// Global->LDS path (GLOBAL_LOAD_ASYNC_TO_LDS_B128 / ASYNCcnt) so the 2-WG/WGP
// occupancy (151KB LDS) does not expose HBM latency.
// Double-eigh in the reference collapses to one Jacobi eigh per sample
// (clamped reconstruction preserves eigenvectors; Wb Wb^T = I).
// ---------------------------------------------------------------------------

typedef __attribute__((ext_vector_type(16))) _Float16 v16h;
typedef __attribute__((ext_vector_type(8)))  float    v8f;

#define NSAMP  2304
#define CH     22
#define T_IN   1000
#define TP     250
#define NP     256     // padded time (250 -> 256)
#define OC     64
#define KW     13
#define KK     286     // 22*13
#define KKP    288     // padded K for conv GEMM
#define RR     32
#define XDS    272     // xd row stride in halves (t index -6..265, stored +6)
#define XCS    96      // centered-signal K padding (84 -> 96)
#define CST    72      // row stride (halves) for C / T1 scratch
#define BN_EPSF 1e-5f

// LDS layout (bytes). Phase A (conv): W | xd | im2col | s  = 151232 B
#define OFF_W    0        // 64*288 f16 = 36864
#define OFF_XD   36864    // 22*272 f16 = 11968
#define OFF_COL  48832    // 64*288 f16 = 36864
#define OFF_S    85696    // 64*256 f32 = 65536
#define SMEM_SZ  151232
// Raw-x async staging aliases [OFF_COL, SMEM_SZ): 22*1000*4 = 88000 B
// (col/s are dead until after downsampling).
#define OFF_RAW  OFF_COL
// Phase B aliases [0, 85696) (W/xd/col dead after conv):
#define OFF_WB   0        // 32*64  f16 = 4096
#define OFF_C    4096     // 64*72  f16 = 9216
#define OFF_T1   13312    // 32*72  f16 = 4608
#define OFF_XC   17920    // 64*96  f16 = 12288
#define OFF_Y    30208    // 32*32  f32 = 4096
#define OFF_RED  34304    // 65 f32 reduction scratch

// ---- CDNA5 async Global->LDS copy (CDNA5 ISA) -------
// Per-lane: LDS[vdst_lds_off] = MEM[vaddr..vaddr+15]; tracked by ASYNCcnt.
__device__ __forceinline__ void async_load_b128(unsigned int lds_off,
                                                const void* gaddr) {
  asm volatile("global_load_async_to_lds_b128 %0, %1, off"
               :: "v"(lds_off), "v"(gaddr)
               : "memory");
}
__device__ __forceinline__ void wait_asynccnt0() {
  asm volatile("s_wait_asynccnt 0" ::: "memory");
}

// ---- WMMA operand loaders (contiguous k = 16*(lane/16)+e, same order for A and B) ----
// A (16x32, M=lane%16): halves e=0..15 -> K = 16*(lane/16)+e (matches load_B)
__device__ __forceinline__ v16h load_A(const _Float16* __restrict__ base,
                                       int stride, int m0, int k0) {
  const int lane = threadIdx.x & 31;
  const _Float16* r0 = base + (m0 + (lane & 15)) * stride + k0 + ((lane >> 4) << 4);
  union { v16h v; _Float16 h[16]; } u;
#pragma unroll
  for (int e = 0; e < 16; ++e) u.h[e] = r0[e];
  return u.v;
}
// B (32x16, N=lane%16): halves e=0..15 -> K = 16*(lane/16)+e.
// Caller passes a source where element (n,k) lives at base + n*stride + k,
// i.e. the k-run is contiguous per lane (one ds_load_b128 pair).
__device__ __forceinline__ v16h load_B(const _Float16* __restrict__ base,
                                       int stride, int n0, int k0) {
  const int lane = threadIdx.x & 31;
  const _Float16* r0 = base + (n0 + (lane & 15)) * stride + k0 + ((lane >> 4) << 4);
  union { v16h v; _Float16 h[16]; } u;
#pragma unroll
  for (int e = 0; e < 16; ++e) u.h[e] = r0[e];
  return u.v;
}
__device__ __forceinline__ v8f wmma_h(v16h a, v16h b, v8f c) {
  return __builtin_amdgcn_wmma_f32_16x16x32_f16(false, a, false, b, (short)0, c,
                                                false, false);
}

// ===========================================================================
// K1: per sample: async-stage x -> downsample -> conv(WMMA) -> BN ->
//     3x{center, cov(WMMA), trace, Wb*C (WMMA), *Wb^T (WMMA)} -> y (32x32).
// ===========================================================================
__global__ __launch_bounds__(256) void rmcnn_fused_front_kernel(
    const float* __restrict__ x,      // (2304, 22, 1000)
    const float* __restrict__ convw,  // (64, 286)
    const float* __restrict__ convb,  // (64)
    const float* __restrict__ bng, const float* __restrict__ bnb,
    const float* __restrict__ bnm, const float* __restrict__ bnv,
    const float* __restrict__ bimap,  // (32, 64)
    float* __restrict__ yws)          // (2304, 32, 32)
{
  __shared__ __align__(16) unsigned char smem[SMEM_SZ];
  __shared__ float bnscale[64];
  __shared__ float bnshift[64];

  _Float16* W_l   = reinterpret_cast<_Float16*>(smem + OFF_W);
  _Float16* xd_l  = reinterpret_cast<_Float16*>(smem + OFF_XD);
  _Float16* col_l = reinterpret_cast<_Float16*>(smem + OFF_COL);
  float*    s_l   = reinterpret_cast<float*>(smem + OFF_S);
  _Float16* Wb_l  = reinterpret_cast<_Float16*>(smem + OFF_WB);
  _Float16* C_l   = reinterpret_cast<_Float16*>(smem + OFF_C);
  _Float16* T1_l  = reinterpret_cast<_Float16*>(smem + OFF_T1);
  _Float16* xc_l  = reinterpret_cast<_Float16*>(smem + OFF_XC);
  float*    y_l   = reinterpret_cast<float*>(smem + OFF_Y);
  float*    redf  = reinterpret_cast<float*>(smem + OFF_RED);

  const int b   = blockIdx.x;
  const int tid = threadIdx.x;

  // ---- Issue async Global->LDS copy of this sample's raw x (88 KB) first,
  //      so it overlaps weight staging / BN precompute below. ----
  {
    const char* gsrc = reinterpret_cast<const char*>(x + (size_t)b * CH * T_IN);
    const unsigned int lbase = (unsigned int)(uintptr_t)(smem + OFF_RAW);
    for (int i = tid; i < (CH * T_IN * 4) / 16; i += 256)  // 5500 x b128
      async_load_b128(lbase + i * 16, gsrc + (size_t)i * 16);
  }

  // ---- Phase A0: zero xd pad, stage conv weights as f16, BN precompute ----
  for (int i = tid; i < CH * XDS; i += 256) xd_l[i] = (_Float16)0.f;
  for (int i = tid; i < OC * KKP; i += 256) {
    const int kk = i % KKP;
    const int oc = i / KKP;
    const float v = (kk < KK) ? convw[oc * KK + kk] : 0.f;
    W_l[i] = (_Float16)v;
  }
  if (tid < 64) {
    const float inv = bng[tid] * rsqrtf(bnv[tid] + BN_EPSF);
    bnscale[tid] = inv;
    bnshift[tid] = bnb[tid] - bnm[tid] * inv + convb[tid] * inv;
  }
  wait_asynccnt0();   // raw x resident in LDS
  __syncthreads();

  // ---- Phase A1: downsample x4 from LDS-staged raw x, f16 with +6 halo ----
  const float* raw = reinterpret_cast<const float*>(smem + OFF_RAW);
  for (int i = tid; i < CH * TP; i += 256) {
    const int c = i / TP, t = i - c * TP;
    const float4 v4 = reinterpret_cast<const float4*>(raw + c * T_IN)[t];
    xd_l[c * XDS + 6 + t] = (_Float16)(0.25f * (v4.x + v4.y + v4.z + v4.w));
  }
  __syncthreads();

  // ---- Phase A2: conv as im2col GEMM, M=64, K=288, N in 4 chunks of 64 ----
  const int w = tid >> 5;  // wave id (wave32), uniform per wave
  for (int nc0 = 0; nc0 < NP; nc0 += 64) {
    for (int i = tid; i < 64 * KKP; i += 256) {
      const int tl = i / KKP, kk = i - tl * KKP;
      _Float16 v = (_Float16)0.f;
      if (kk < KK) {
        const int c = kk / KW, k2 = kk - c * KW;
        v = xd_l[c * XDS + (nc0 + tl) + k2];  // xd[c][t+k2-6] with +6 halo
      }
      col_l[i] = v;
    }
    __syncthreads();
#pragma unroll
    for (int t2 = 0; t2 < 2; ++t2) {
      const int tile = w * 2 + t2;
      const int mt = tile >> 2, nt = tile & 3;
      v8f acc = {0.f, 0.f, 0.f, 0.f, 0.f, 0.f, 0.f, 0.f};
#pragma unroll
      for (int ks = 0; ks < 9; ++ks)
        acc = wmma_h(load_A(W_l, KKP, mt * 16, ks * 32),
                     load_B(col_l, KKP, nt * 16, ks * 32), acc);
      const int lane = tid & 31;
      const int n = nc0 + nt * 16 + (lane & 15);
      const int mb = mt * 16 + ((lane >> 4) << 3);
#pragma unroll
      for (int r = 0; r < 8; ++r) {
        const int m = mb + r;
        s_l[m * NP + n] = acc[r] * bnscale[m] + bnshift[m];
      }
    }
    __syncthreads();
  }

  // ---- Phase B: covariance + bimap per time patch ----
  for (int i = tid; i < RR * OC; i += 256) Wb_l[i] = (_Float16)bimap[i];
  for (int i = tid; i < RR * RR; i += 256) y_l[i] = 0.f;
  __syncthreads();

  const int offs[3] = {0, 84, 167};
  const int lens[3] = {84, 83, 83};
  for (int blk = 0; blk < 3; ++blk) {
    const int off = offs[blk], len = lens[blk];
    // per-channel mean over the patch
    if (tid < 64) {
      float sum = 0.f;
      for (int jj = 0; jj < len; ++jj) sum += s_l[tid * NP + off + jj];
      redf[tid] = sum / (float)len;
    }
    __syncthreads();
    // centered signal (f16), K padded 84->96 with zeros
    for (int i = tid; i < 64 * XCS; i += 256) {
      const int ch = i / XCS, jj = i - ch * XCS;
      const float v = (jj < len) ? (s_l[ch * NP + off + jj] - redf[ch]) : 0.f;
      xc_l[i] = (_Float16)v;
    }
    __syncthreads();
    // C = xc * xc^T (64x64, K=96). 1/(l-1) cancels against trace-normalize.
#pragma unroll
    for (int t2 = 0; t2 < 2; ++t2) {
      const int tile = w * 2 + t2;
      const int mt = tile >> 2, nt = tile & 3;
      v8f acc = {0.f, 0.f, 0.f, 0.f, 0.f, 0.f, 0.f, 0.f};
#pragma unroll
      for (int ks = 0; ks < 3; ++ks)
        acc = wmma_h(load_A(xc_l, XCS, mt * 16, ks * 32),
                     load_B(xc_l, XCS, nt * 16, ks * 32), acc);
      const int lane = tid & 31;
      const int n = nt * 16 + (lane & 15);
      const int mb = mt * 16 + ((lane >> 4) << 3);
#pragma unroll
      for (int r = 0; r < 8; ++r) C_l[(mb + r) * CST + n] = (_Float16)acc[r];
    }
    __syncthreads();
    // trace(C)
    if (tid < 64) redf[tid] = (float)C_l[tid * CST + tid];
    __syncthreads();
    if (tid == 0) {
      float tr = 0.f;
      for (int i = 0; i < 64; ++i) tr += redf[i];
      redf[64] = 1.0f / (3.0f * tr);  // (cov/trace)/BLOCKS
    }
    __syncthreads();
    // T1 = Wb (32x64) * C (64x64); B-operand uses C row n (C symmetric)
    {
      const int mt = w >> 2, nt = w & 3;
      v8f acc = {0.f, 0.f, 0.f, 0.f, 0.f, 0.f, 0.f, 0.f};
#pragma unroll
      for (int ks = 0; ks < 2; ++ks)
        acc = wmma_h(load_A(Wb_l, OC, mt * 16, ks * 32),
                     load_B(C_l, CST, nt * 16, ks * 32), acc);
      const int lane = tid & 31;
      const int n = nt * 16 + (lane & 15);
      const int mb = mt * 16 + ((lane >> 4) << 3);
#pragma unroll
      for (int r = 0; r < 8; ++r) T1_l[(mb + r) * CST + n] = (_Float16)acc[r];
    }
    __syncthreads();
    // P = T1 (32x64) * Wb^T; B-operand = Wb row n (k-contiguous). y += P*scale
    if (w < 4) {
      const int mt = w >> 1, nt = w & 1;
      v8f acc = {0.f, 0.f, 0.f, 0.f, 0.f, 0.f, 0.f, 0.f};
#pragma unroll
      for (int ks = 0; ks < 2; ++ks)
        acc = wmma_h(load_A(T1_l, CST, mt * 16, ks * 32),
                     load_B(Wb_l, OC, nt * 16, ks * 32), acc);
      const float sc = redf[64];
      const int lane = tid & 31;
      const int n = nt * 16 + (lane & 15);
      const int mb = mt * 16 + ((lane >> 4) << 3);
#pragma unroll
      for (int r = 0; r < 8; ++r) y_l[(mb + r) * RR + n] += acc[r] * sc;
    }
    __syncthreads();
  }

  if (tid < RR) y_l[tid * RR + tid] += 1e-5f;  // + eps*I (Wb Wb^T = I)
  __syncthreads();
  for (int i = tid; i < RR * RR; i += 256) yws[(size_t)b * (RR * RR) + i] = y_l[i];
}

// ===========================================================================
// K2: one wave per sample: cyclic Jacobi eigh of y (32x32), clamp+log eigs,
//     logm = V diag(log max(s,eps)) V^T, emit scaled upper-tri -> fe.
// ===========================================================================
__global__ __launch_bounds__(32) void rmcnn_eig_logm_kernel(
    const float* __restrict__ yws, float* __restrict__ fe)
{
  __shared__ float A[32 * 33];
  __shared__ float V[32 * 33];
  __shared__ float evl[32];
  const int b = blockIdx.x;
  const int j = threadIdx.x;  // lane 0..31 (wave32)

  for (int i = 0; i < 32; ++i) {
    A[i * 33 + j] = yws[(size_t)b * 1024 + i * 32 + j];
    V[i * 33 + j] = (i == j) ? 1.f : 0.f;
  }
  __syncthreads();

  for (int sweep = 0; sweep < 8; ++sweep) {
    for (int p = 0; p < 31; ++p) {
      for (int q = p + 1; q < 32; ++q) {
        const float app = A[p * 33 + p];
        const float aqq = A[q * 33 + q];
        const float apq = A[p * 33 + q];
        float c, s;
        if (fabsf(apq) > 1e-12f) {
          const float theta = 0.5f * (aqq - app) / apq;
          const float t = (theta >= 0.f ? 1.f : -1.f) /
                          (fabsf(theta) + sqrtf(1.f + theta * theta));
          c = rsqrtf(1.f + t * t);
          s = t * c;
        } else {
          c = 1.f; s = 0.f;
        }
        __syncthreads();
        { // rows p,q  (A <- G A)
          const float apj = A[p * 33 + j], aqj = A[q * 33 + j];
          A[p * 33 + j] = c * apj - s * aqj;
          A[q * 33 + j] = s * apj + c * aqj;
        }
        __syncthreads();
        { // cols p,q  (A <- A G^T), accumulate V <- V G^T
          const float ajp = A[j * 33 + p], ajq = A[j * 33 + q];
          A[j * 33 + p] = c * ajp - s * ajq;
          A[j * 33 + q] = s * ajp + c * ajq;
          const float vjp = V[j * 33 + p], vjq = V[j * 33 + q];
          V[j * 33 + p] = c * vjp - s * vjq;
          V[j * 33 + q] = s * vjp + c * vjq;
        }
        __syncthreads();
      }
    }
  }

  evl[j] = logf(fmaxf(A[j * 33 + j], 1e-4f));  // clamp (EPS_REC) then log
  __syncthreads();

  float vw[32];
#pragma unroll
  for (int k = 0; k < 32; ++k) vw[k] = V[j * 33 + k] * evl[k];

  int base = 0;
  for (int i = 0; i < 32; ++i) {
    float acc = 0.f;
#pragma unroll
    for (int k = 0; k < 32; ++k) acc += V[i * 33 + k] * vw[k];
    if (j >= i) {
      const float sc = (j == i) ? 1.f : 1.41421356237f;
      fe[(size_t)b * 528 + base + (j - i)] = acc * sc;
    }
    base += 32 - i;
  }
}

// ===========================================================================
// K3: logits = fe (256x4752) @ linear_w^T (4752x4) + b
// ===========================================================================
__global__ __launch_bounds__(128) void rmcnn_logits_kernel(
    const float* __restrict__ fe, const float* __restrict__ lw,
    const float* __restrict__ lb, float* __restrict__ out)
{
  __shared__ float red[128];
  const int row = blockIdx.x;
  const int tid = threadIdx.x;
  for (int cls = 0; cls < 4; ++cls) {
    float acc = 0.f;
    for (int k = tid; k < 4752; k += 128)
      acc += fe[(size_t)row * 4752 + k] * lw[cls * 4752 + k];
    red[tid] = acc;
    __syncthreads();
    for (int s = 64; s > 0; s >>= 1) {
      if (tid < s) red[tid] += red[tid + s];
      __syncthreads();
    }
    if (tid == 0) out[row * 4 + cls] = red[0] + lb[cls];
    __syncthreads();
  }
}

// ===========================================================================
extern "C" void kernel_launch(void* const* d_in, const int* in_sizes, int n_in,
                              void* d_out, int out_size, void* d_ws, size_t ws_size,
                              hipStream_t stream) {
  const float* x     = (const float*)d_in[0];  // (256,9,22,1000)
  const float* convw = (const float*)d_in[1];  // (64,1,22,13)
  const float* convb = (const float*)d_in[2];
  const float* bng   = (const float*)d_in[3];
  const float* bnb   = (const float*)d_in[4];
  const float* bnm   = (const float*)d_in[5];
  const float* bnv   = (const float*)d_in[6];
  const float* bimap = (const float*)d_in[7];  // (32,64)
  const float* lw    = (const float*)d_in[8];  // (4,4752)
  const float* lb    = (const float*)d_in[9];

  float* fe     = (float*)d_out;                    // 2304*528 = 256*4752
  float* logits = (float*)d_out + (size_t)2304 * 528;
  float* yws    = (float*)d_ws;                     // 2304*1024 f32 = 9.4 MB

  rmcnn_fused_front_kernel<<<NSAMP, 256, 0, stream>>>(
      x, convw, convb, bng, bnb, bnm, bnv, bimap, yws);
  rmcnn_eig_logm_kernel<<<NSAMP, 32, 0, stream>>>(yws, fe);
  rmcnn_logits_kernel<<<256, 128, 0, stream>>>(fe, lw, lb, logits);
}
